// ConvolutionalAttention_16028817949217
// MI455X (gfx1250) — matrix operand order repaired and checked by reading, not compiled, not run on hardware
//
#include <hip/hip_runtime.h>
#include <hip/hip_bf16.h>
#include <math.h>

#define B_    16
#define C_    64
#define H_    192
#define W_    192
#define HW    (H_*W_)
#define LK    13
#define NCH   169          // 13*13 K-chunks for the large-kernel conv
#define HT    8
#define WT    32
#define TPIX  (HT*WT)      // 256 output pixels / workgroup
#define TIN_H (HT+12)      // 20 (halo 6 each side)
#define TIN_W (WT+12)      // 44
#define CH_PAD 40          // padded ci stride (bf16 elems): 80B row -> 16B-aligned subchunks
#define PH    (H_+12)      // 204: zero-bordered prepacked tensor height
#define PW    (W_+12)      // 204

typedef __attribute__((ext_vector_type(16))) __bf16       v16bf;
typedef __attribute__((ext_vector_type(8)))  float        v8f;
typedef __attribute__((ext_vector_type(4)))  unsigned int u32x4;
typedef __attribute__((ext_vector_type(8)))  int          i32x8;
typedef __attribute__((ext_vector_type(4)))  int          i32x4;

union F16x16 { v16bf v; u32x4 q[2]; unsigned short s[16]; };

__device__ __forceinline__ unsigned short f2bf(float f) {
  unsigned int u = __float_as_uint(f);
  u += 0x7fffu + ((u >> 16) & 1u);          // round-to-nearest-even
  return (unsigned short)(u >> 16);
}

// ---------------- pooled = mean(x1, H, W) : one block per (b, c<32) ----------------
__global__ void pool_k(const float* __restrict__ x, float* __restrict__ pooled) {
  int bc = blockIdx.x;                       // 0..511
  int b = bc >> 5, c = bc & 31;
  const float* p = x + ((size_t)b * C_ + c) * HW;
  float s = 0.f;
  for (int i = threadIdx.x; i < HW; i += 256) s += p[i];
  __shared__ float red[256];
  red[threadIdx.x] = s; __syncthreads();
  for (int off = 128; off > 0; off >>= 1) {
    if (threadIdx.x < off) red[threadIdx.x] += red[threadIdx.x + off];
    __syncthreads();
  }
  if (threadIdx.x == 0) pooled[bc] = red[0] * (1.f / (float)HW);
}

// ---------------- tiny MLP: hid = gelu(pooled@w1^T+b1); dk = hid@w2^T+b2 ----------------
__global__ void mlp_k(const float* __restrict__ pooled, const float* __restrict__ w1,
                      const float* __restrict__ b1, const float* __restrict__ w2,
                      const float* __restrict__ b2, float* __restrict__ dk) {
  __shared__ float sp[512];
  __shared__ float sh[256];
  int tid = threadIdx.x;                     // 288 threads
  for (int i = tid; i < 512; i += 288) sp[i] = pooled[i];
  __syncthreads();
  if (tid < 256) {
    int b = tid >> 4, j = tid & 15;
    float a = b1[j];
    for (int c = 0; c < 32; ++c) a += sp[b * 32 + c] * w1[j * 32 + c];
    sh[tid] = 0.5f * a * (1.f + erff(a * 0.70710678118654752f));  // exact gelu
  }
  __syncthreads();
  {
    int o = tid;                             // 0..287
    float bb = b2[o];
    for (int b = 0; b < 16; ++b) {
      float a = bb;
      for (int j = 0; j < 16; ++j) a += sh[b * 16 + j] * w2[o * 16 + j];
      dk[b * 288 + o] = a;                   // dk[b][co][tap], tap fastest
    }
  }
}

// ---------------- reorder lk_filter (co,ci,kh,kw) f32 -> fr[co][tap][ci] bf16 ----------------
__global__ void fprep_k(const float* __restrict__ lkf, unsigned short* __restrict__ fr) {
  int f = blockIdx.x * 256 + threadIdx.x;    // 32*169*32 = 173056 total (exact grid)
  int ci = f & 31;
  int chunk = (f >> 5) % NCH;
  int co = f / (NCH * 32);
  fr[f] = f2bf(lkf[(co * 32 + ci) * NCH + chunk]);
}

// ---- prepack x1: f32 NCHW -> zero-bordered channels-last bf16 xp[b][204][204][32] ----
__global__ void prepack_k(const float* __restrict__ x, unsigned short* __restrict__ xp) {
  __shared__ unsigned short row[PW * 32];    // one output row, 13056 B
  int bh = blockIdx.x;                       // 0..16*204-1
  int b = bh / PH, hp = bh % PH;
  int h = hp - 6;
  const float* xb = x + (size_t)b * C_ * HW;
  for (int ci = 0; ci < 32; ++ci) {
    const float* xr = xb + (size_t)ci * HW + (size_t)h * W_;
    for (int p = threadIdx.x; p < PW; p += 256) {
      int w = p - 6;
      float f = 0.f;
      if ((unsigned)h < (unsigned)H_ && (unsigned)w < (unsigned)W_) f = xr[w];
      row[p * 32 + ci] = f2bf(f);
    }
  }
  __syncthreads();
  unsigned int* dst = (unsigned int*)(xp + (size_t)bh * PW * 32);
  const unsigned int* src = (const unsigned int*)row;
  for (int d = threadIdx.x; d < PW * 16; d += 256) dst[d] = src[d];
}

// ---------------- out[:,32:64] = x[:,32:64] passthrough (float4) ----------------
__global__ void copy_k(const float* __restrict__ x, float* __restrict__ out) {
  size_t i4 = (size_t)blockIdx.x * 256 + threadIdx.x;   // over count/4 (exact grid)
  size_t idx = i4 * 4;
  size_t per_b = (size_t)32 * HW;
  size_t b = idx / per_b, r = idx % per_b;
  size_t off = (b * C_ + 32) * HW + r;
  *(float4*)(out + off) = *(const float4*)(x + off);
}

// ---------------- fused lk(13x13 dense) + dyn(3x3 depthwise) via WMMA implicit GEMM ----------------
union __align__(16) SMEM {
  unsigned short tin[TIN_H * TIN_W * CH_PAD];  // 70400 B, bf16 channels-last input tile
  float cst[TPIX * 33];                        // 33792 B, C-tile transpose staging
};

__launch_bounds__(256)
__global__ void lkdyn_k(const unsigned short* __restrict__ xp,
                        const unsigned short* __restrict__ fr,
                        const float* __restrict__ dk, float* __restrict__ out) {
  __shared__ SMEM sm;
  const int tid  = threadIdx.x;
  const int lane = tid & 31;
  const int wave = tid >> 5;
  const int gsel = lane >> 4;          // which half-wave (K-split of A, K16-split of B)
  const int b  = blockIdx.z;
  const int h0 = blockIdx.y * HT;
  const int w0 = blockIdx.x * WT;

  // ---- TDM: DMA the (20 x 44 x 32ch) bf16 tile into LDS with 64B->80B row padding ----
  if (wave == 0) {
    unsigned long long ga = (unsigned long long)(uintptr_t)xp +
        (((size_t)b * PH + h0) * PW + w0) * 32ull * 2ull;   // tile start (always in-bounds)
    unsigned int ldsa = (unsigned int)(uintptr_t)&sm.tin[0];

    u32x4 g0;
    g0.x = 1u;                                              // count=1 valid descriptor
    g0.y = ldsa;                                            // lds_addr (bytes)
    g0.z = (unsigned int)ga;                                // global_addr[31:0]
    g0.w = (unsigned int)((ga >> 32) & 0x1FFFFFFu) | (2u << 30);  // addr[56:32] | type=2

    i32x8 g1 = (i32x8)0;
    g1[0] = (1 << 16)      // data_size = 2 bytes
          | (1 << 20)      // pad_enable
          | (3 << 22)      // pad_interval: every 16 dwords (one 32ch pixel)
          | (3 << 25);     // pad_amount: 4 dwords (16 B) -> 80 B pixel stride
    g1[1] = (int)(((unsigned)(PW * 32) & 0xFFFFu) << 16);   // tensor_dim0[15:0]
    g1[2] = (int)(((unsigned)(PW * 32) >> 16) | ((unsigned)PH << 16)); // dim0[31:16] | dim1[15:0]
    g1[3] = (int)((unsigned)(TIN_W * 32) << 16);            // tile_dim0 = 1408 elems
    g1[4] = TIN_H;                                          // tile_dim1 = 20 rows
    g1[5] = PW * 32;                                        // tensor_dim0_stride
    i32x4 gz = (i32x4)0;                                    // 2D: groups 2/3 unused

#if defined(__clang_major__) && (__clang_major__ >= 23)
    __builtin_amdgcn_tensor_load_to_lds(g0, g1, gz, gz, (i32x8)0, 0);
#else
    __builtin_amdgcn_tensor_load_to_lds(g0, g1, gz, gz, 0);
#endif
    __builtin_amdgcn_s_wait_tensorcnt(0);
  }

  // per-lane dynamic-kernel taps (diagonal channel co == lane)
  unsigned short dkb[9];
  {
    const float* d = dk + ((size_t)b * 32 + lane) * 9;
#pragma unroll
    for (int t = 0; t < 9; ++t) dkb[t] = f2bf(d[t]);
  }

  __syncthreads();

  // per-lane A-fragment pixel coords for the two M-subtiles this wave owns
  int ay[2], ax[2];
#pragma unroll
  for (int ms = 0; ms < 2; ++ms) {
    int m = wave * 32 + ms * 16 + (lane & 15);
    ay[ms] = m >> 5; ax[ms] = m & 31;
  }

  v8f acc[2][2] = {};   // [msub][nsub] 16x16 f32 accumulators

  // ---- 169 K-chunks: dense 13x13 conv, K=32 channels per tap ----
  int kh = 0, kw = 0;
  for (int chunk = 0; chunk < NCH; ++chunk) {
    F16x16 af[2], bfr[2];
#pragma unroll
    for (int ms = 0; ms < 2; ++ms) {
      int pid = (ay[ms] + kh) * TIN_W + (ax[ms] + kw);
      const u32x4* pa = (const u32x4*)&sm.tin[pid * CH_PAD + 16 * gsel];
      af[ms].q[0] = pa[0];             // K = 16g..16g+7
      af[ms].q[1] = pa[1];             // K = 16g+8..16g+15
    }
#pragma unroll
    for (int ns = 0; ns < 2; ++ns) {
      int co = ns * 16 + (lane & 15);
      const u32x4* pb = (const u32x4*)&fr[(size_t)(co * NCH + chunk) * 32 + 16 * gsel];
      bfr[ns].q[0] = pb[0];            // K = 16g..16g+15 for column co
      bfr[ns].q[1] = pb[1];
    }
#pragma unroll
    for (int ms = 0; ms < 2; ++ms)
#pragma unroll
      for (int ns = 0; ns < 2; ++ns)
        acc[ms][ns] = __builtin_amdgcn_wmma_f32_16x16x32_bf16(
            false, af[ms].v, false, bfr[ns].v, (short)0, acc[ms][ns], false, false);
    if (++kw == LK) { kw = 0; ++kh; }
  }

  // ---- 9 extra K-chunks: dynamic depthwise 3x3 as diagonal-B GEMM ----
#pragma unroll
  for (int t = 0; t < 9; ++t) {
    const int dh = 5 + t / 3, dw = 5 + t % 3;   // tap offset inside halo-6 frame
    F16x16 af[2], bd[2];
#pragma unroll
    for (int ms = 0; ms < 2; ++ms) {
      int pid = (ay[ms] + dh) * TIN_W + (ax[ms] + dw);
      const u32x4* pa = (const u32x4*)&sm.tin[pid * CH_PAD + 16 * gsel];
      af[ms].q[0] = pa[0];
      af[ms].q[1] = pa[1];
    }
#pragma unroll
    for (int ns = 0; ns < 2; ++ns) {
#pragma unroll
      for (int e = 0; e < 16; ++e)
        bd[ns].s[e] = ((gsel == ns) && ((lane & 15) == e)) ? dkb[t] : (unsigned short)0;
    }
#pragma unroll
    for (int ms = 0; ms < 2; ++ms)
#pragma unroll
      for (int ns = 0; ns < 2; ++ns)
        acc[ms][ns] = __builtin_amdgcn_wmma_f32_16x16x32_bf16(
            false, af[ms].v, false, bd[ns].v, (short)0, acc[ms][ns], false, false);
  }

  __syncthreads();   // tile buffer done; reuse LDS as f32 C staging

#pragma unroll
  for (int ms = 0; ms < 2; ++ms)
#pragma unroll
    for (int ns = 0; ns < 2; ++ns) {
      int co = ns * 16 + (lane & 15);
#pragma unroll
      for (int r = 0; r < 8; ++r) {
        int m = wave * 32 + ms * 16 + r + 8 * gsel;   // C layout: lane-half -> M+8
        sm.cst[m * 33 + co] = acc[ms][ns][r];
      }
    }
  __syncthreads();

  // coalesced NCHW stores: tid sweeps the 256 pixels (w fastest) per channel
  float* ob = out + (size_t)b * C_ * HW;
  int yy = tid >> 5, xx = tid & 31;
#pragma unroll 4
  for (int co = 0; co < 32; ++co) {
    float v = sm.cst[tid * 33 + co];
    ob[(size_t)co * HW + (size_t)(h0 + yy) * W_ + (w0 + xx)] = v;
  }
}

// ---------------- launch ----------------
extern "C" void kernel_launch(void* const* d_in, const int* in_sizes, int n_in,
                              void* d_out, int out_size, void* d_ws, size_t ws_size,
                              hipStream_t stream) {
  const float* x   = (const float*)d_in[0];
  const float* lkf = (const float*)d_in[1];
  const float* w1  = (const float*)d_in[2];
  const float* b1  = (const float*)d_in[3];
  const float* w2  = (const float*)d_in[4];
  const float* b2  = (const float*)d_in[5];
  float* out = (float*)d_out;

  char* ws = (char*)d_ws;
  float*          pooled = (float*)(ws);                    //   512 f32
  float*          dk     = (float*)(ws + 2048);             //  4608 f32 (16x32x9)
  unsigned short* fr     = (unsigned short*)(ws + 20480);   // 173056 bf16 (~346 KB)
  unsigned short* xp     = (unsigned short*)(ws + 524288);  // 16*204*204*32 bf16 (~42.6 MB)

  pool_k   <<<512, 256, 0, stream>>>(x, pooled);
  mlp_k    <<<1, 288, 0, stream>>>(pooled, w1, b1, w2, b2, dk);
  fprep_k  <<<(32 * NCH * 32) / 256, 256, 0, stream>>>(lkf, fr);
  prepack_k<<<B_ * PH, 256, 0, stream>>>(x, xp);
  copy_k   <<<(int)(((size_t)B_ * 32 * HW / 4) / 256), 256, 0, stream>>>(x, out);
  lkdyn_k  <<<dim3(W_ / WT, H_ / HT, B_), 256, 0, stream>>>(xp, fr, dk, out);
}
